// MambaBottleneck_58342835749262
// MI455X (gfx1250) — matrix operand order repaired and checked by reading, not compiled, not run on hardware
//
#include <hip/hip_runtime.h>
#include <hip/hip_bf16.h>
#include <math.h>

// ---------------- problem constants ----------------
#define BDIM   8
#define CDIM   512
#define HH     24
#define WWDIM  24
#define LSEQ   (HH*WWDIM)          // 576
#define MROWS  (BDIM*LSEQ)         // 4608
#define DINNER 1024
#define DTRANK 32
#define DSTATE 16
#define KCONV  (9*CDIM)            // 4608
#define PADHW  26
#define PADAREA (PADHW*PADHW)      // 676

// ---------------- WMMA types ----------------
typedef __attribute__((ext_vector_type(16))) __bf16          v16bf;
typedef __attribute__((ext_vector_type(16))) unsigned short  v16us;
typedef __attribute__((ext_vector_type(8)))  unsigned short  v8us;
typedef __attribute__((ext_vector_type(8)))  float           v8f;

union U16V { v16us v; v8us h[2]; unsigned short s[16]; };

__device__ __forceinline__ unsigned short f2bf(float f) {
  unsigned int u = __float_as_uint(f);
  unsigned int r = u + 0x7FFFu + ((u >> 16) & 1u);   // round-to-nearest-even
  return (unsigned short)(r >> 16);
}
__device__ __forceinline__ float bf2f(unsigned short s) {
  return __uint_as_float(((unsigned int)s) << 16);
}

// ---------------- epilogue modes ----------------
#define EPI_CONV_IN  0   // +bias, BN, ReLU -> seq f32 (b,l,c)
#define EPI_BF16     1   // -> bf16 row-major
#define EPI_XPROJ    2   // -> f32 (M,64); cols<32 mirrored to bf16 dt buffer
#define EPI_SOFTPLUS 3   // +bias, softplus -> f32
#define EPI_RESID    4   // += into seq f32
#define EPI_CONV_OUT 5   // +bias, BN, ReLU -> NCHW f32 d_out

__device__ __forceinline__ void epilogue_store(
    int mode, int m, int n, int N, float val,
    const float* __restrict__ bias, const float* __restrict__ gamma,
    const float* __restrict__ beta,
    float* __restrict__ outF, unsigned short* __restrict__ outBF)
{
  switch (mode) {
    case EPI_CONV_IN: {
      float v = val + bias[n];
      v = v * (gamma[n] * rsqrtf(1.0f + 1e-5f)) + beta[n];
      v = v > 0.f ? v : 0.f;
      outF[(size_t)m * CDIM + n] = v;
    } break;
    case EPI_BF16:
      outBF[(size_t)m * N + n] = f2bf(val);
      break;
    case EPI_XPROJ:
      outF[(size_t)m * 64 + n] = val;
      if (n < DTRANK) outBF[(size_t)m * DTRANK + n] = f2bf(val);
      break;
    case EPI_SOFTPLUS: {
      float v = val + bias[n];
      float sp = (v > 20.f) ? v : logf(1.f + __expf(v));
      outF[(size_t)m * N + n] = sp;
    } break;
    case EPI_RESID:
      outF[(size_t)m * CDIM + n] += val;
      break;
    case EPI_CONV_OUT: {
      float v = val + bias[n];
      v = v * (gamma[n] * rsqrtf(1.0f + 1e-5f)) + beta[n];
      v = v > 0.f ? v : 0.f;
      int b = m / LSEQ, l = m % LSEQ;
      outF[((size_t)b * CDIM + n) * LSEQ + l] = v;
    } break;
  }
}

// ---------------- dense bf16 WMMA GEMM: Y = A(MxK) * Wt(NxK)^T ----------------
// One wave per 32x64 output tile; 2x4 fragments of v_wmma_f32_16x16x32_bf16.
__global__ __launch_bounds__(32)
void gemm_bf16_wmma(const unsigned short* __restrict__ A,
                    const unsigned short* __restrict__ Wt,
                    int M, int N, int K, int mode,
                    const float* __restrict__ bias,
                    const float* __restrict__ gamma,
                    const float* __restrict__ beta,
                    float* __restrict__ outF,
                    unsigned short* __restrict__ outBF)
{
  const int lane  = threadIdx.x;
  const int m0    = blockIdx.y * 32;
  const int n0    = blockIdx.x * 64;
  const int rlo   = lane & 15;
  const int khalf = lane >> 4;   // 0 or 1

  v8f acc[2][4] = {};

  for (int k0 = 0; k0 < K; k0 += 32) {
    U16V a[2], bmat[4];
    // A fragments: 16x32 per frag; lane half picks K base (0..7/16..23 vs 8..15/24..31)
    #pragma unroll
    for (int i = 0; i < 2; ++i) {
      const unsigned short* p = A + (size_t)(m0 + i * 16 + rlo) * K + k0 + (khalf << 3);
      a[i].h[0] = *(const v8us*)p;
      a[i].h[1] = *(const v8us*)(p + 16);
    }
    // B fragments: 32x16 per frag; lane = N column, lane half picks K base (0..7/16..23 vs 8..15/24..31)
    #pragma unroll
    for (int j = 0; j < 4; ++j) {
      const unsigned short* q = Wt + (size_t)(n0 + j * 16 + rlo) * K + k0 + (khalf << 3);
      bmat[j].h[0] = *(const v8us*)q;
      bmat[j].h[1] = *(const v8us*)(q + 16);
    }
    if (k0 + 32 < K) {
      __builtin_prefetch(A + (size_t)(m0 + rlo) * K + k0 + 32, 0, 1);
      __builtin_prefetch(Wt + (size_t)(n0 + rlo) * K + k0 + 32, 0, 1);
    }
    #pragma unroll
    for (int i = 0; i < 2; ++i)
      #pragma unroll
      for (int j = 0; j < 4; ++j)
        acc[i][j] = __builtin_amdgcn_wmma_f32_16x16x32_bf16(
            false, __builtin_bit_cast(v16bf, a[i].v),
            false, __builtin_bit_cast(v16bf, bmat[j].v),
            (short)0, acc[i][j], false, false);
  }

  #pragma unroll
  for (int i = 0; i < 2; ++i)
    #pragma unroll
    for (int j = 0; j < 4; ++j)
      #pragma unroll
      for (int v = 0; v < 8; ++v) {
        int m = m0 + i * 16 + v + (khalf << 3);
        int n = n0 + j * 16 + rlo;
        epilogue_store(mode, m, n, N, acc[i][j][v], bias, gamma, beta, outF, outBF);
      }
}

// ---------------- conv3x3 implicit-GEMM via WMMA ----------------
// A row m = (b, h, w); k = p*512 + ci, p = kh*3+kw.
// Padded input is bf16 NHWC (B, 26, 26, 512) so each 8-element sub-fragment
// (fixed tap p, ci contiguous) is a single 16-byte contiguous load.
// Weight reordered (o, p, ci) so B fragments are two contiguous 16-byte loads per lane.
__global__ __launch_bounds__(32)
void conv3x3_gemm_wmma(const unsigned short* __restrict__ pad,
                       const unsigned short* __restrict__ Wt,
                       int mode,
                       const float* __restrict__ bias,
                       const float* __restrict__ gamma,
                       const float* __restrict__ beta,
                       float* __restrict__ outF)
{
  const int lane  = threadIdx.x;
  const int m0    = blockIdx.y * 32;
  const int n0    = blockIdx.x * 64;
  const int rlo   = lane & 15;
  const int khalf = lane >> 4;

  v8f acc[2][4] = {};

  int rowP[2];   // padded-origin pixel index (b*26 + h)*26 + w
  #pragma unroll
  for (int i = 0; i < 2; ++i) {
    int m = m0 + i * 16 + rlo;
    int b = m / LSEQ, l = m % LSEQ;
    int h = l / WWDIM, w = l % WWDIM;
    rowP[i] = (b * PADHW + h) * PADHW + w;
  }

  for (int k0 = 0; k0 < KCONV; k0 += 32) {
    U16V a[2], bmat[4];
    #pragma unroll
    for (int i = 0; i < 2; ++i) {
      #pragma unroll
      for (int hf = 0; hf < 2; ++hf) {
        int k8  = k0 + (khalf << 3) + hf * 16;  // 8 consecutive k share one tap p
        int p   = k8 >> 9;
        int ci0 = k8 & 511;
        int kh  = p / 3, kw = p % 3;
        const unsigned short* src =
            pad + ((size_t)(rowP[i] + kh * PADHW + kw) << 9) + ci0;
        a[i].h[hf] = *(const v8us*)src;
      }
    }
    #pragma unroll
    for (int j = 0; j < 4; ++j) {
      const unsigned short* q = Wt + (size_t)(n0 + j * 16 + rlo) * KCONV + k0 + (khalf << 3);
      bmat[j].h[0] = *(const v8us*)q;
      bmat[j].h[1] = *(const v8us*)(q + 16);
    }
    #pragma unroll
    for (int i = 0; i < 2; ++i)
      #pragma unroll
      for (int j = 0; j < 4; ++j)
        acc[i][j] = __builtin_amdgcn_wmma_f32_16x16x32_bf16(
            false, __builtin_bit_cast(v16bf, a[i].v),
            false, __builtin_bit_cast(v16bf, bmat[j].v),
            (short)0, acc[i][j], false, false);
  }

  #pragma unroll
  for (int i = 0; i < 2; ++i)
    #pragma unroll
    for (int j = 0; j < 4; ++j)
      #pragma unroll
      for (int v = 0; v < 8; ++v) {
        int m = m0 + i * 16 + v + (khalf << 3);
        int n = n0 + j * 16 + rlo;
        epilogue_store(mode, m, n, CDIM, acc[i][j][v], bias, gamma, beta, outF, nullptr);
      }
}

// ---------------- LayerNorm: one wave per token row (512 cols) ----------------
__global__ __launch_bounds__(32)
void layernorm_rows(const float* __restrict__ x,
                    const float* __restrict__ wln, const float* __restrict__ bln,
                    unsigned short* __restrict__ out)
{
  int row  = blockIdx.x;
  int lane = threadIdx.x;
  float vals[16];
  float s = 0.f;
  #pragma unroll
  for (int e = 0; e < 16; ++e) {
    vals[e] = x[(size_t)row * CDIM + e * 32 + lane];
    s += vals[e];
  }
  #pragma unroll
  for (int off = 16; off >= 1; off >>= 1) s += __shfl_xor(s, off, 32);
  float mu = s * (1.f / CDIM);
  float vs = 0.f;
  #pragma unroll
  for (int e = 0; e < 16; ++e) { float d = vals[e] - mu; vs += d * d; }
  #pragma unroll
  for (int off = 16; off >= 1; off >>= 1) vs += __shfl_xor(vs, off, 32);
  float rstd = rsqrtf(vs * (1.f / CDIM) + 1e-5f);
  #pragma unroll
  for (int e = 0; e < 16; ++e) {
    int c = e * 32 + lane;
    out[(size_t)row * CDIM + c] = f2bf((vals[e] - mu) * rstd * wln[c] + bln[c]);
  }
}

// ---------------- causal depthwise conv1d (taps=4) + SiLU ----------------
__global__ void dwconv1d_silu(const unsigned short* __restrict__ xz,
                              const float* __restrict__ cw, const float* __restrict__ cb,
                              unsigned short* __restrict__ xc)
{
  int idx = blockIdx.x * blockDim.x + threadIdx.x;
  if (idx >= MROWS * DINNER) return;
  int d = idx & (DINNER - 1);
  int m = idx >> 10;
  int b = m / LSEQ, t = m % LSEQ;
  float acc = cb[d];
  #pragma unroll
  for (int j = 0; j < 4; ++j) {
    int tt = t - 3 + j;
    if (tt >= 0)
      acc += bf2f(xz[((size_t)(b * LSEQ + tt)) * (2 * DINNER) + d]) * cw[d * 4 + j];
  }
  float y = acc / (1.f + __expf(-acc));   // SiLU
  xc[idx] = f2bf(y);
}

// ---------------- fused selective scan + u*D skip + SiLU(z) gating ----------------
// One thread per (b, d); 16 fp32 states in registers; sequential over L=576.
__global__ void selective_scan_fused(const float* __restrict__ delta,
                                     const unsigned short* __restrict__ xc,
                                     const float* __restrict__ xdbl,
                                     const unsigned short* __restrict__ xz,
                                     const float* __restrict__ a_log,
                                     const float* __restrict__ dskip,
                                     unsigned short* __restrict__ yg)
{
  int idx = blockIdx.x * blockDim.x + threadIdx.x;
  if (idx >= BDIM * DINNER) return;
  int b = idx >> 10;
  int d = idx & (DINNER - 1);

  float Arow[DSTATE];
  #pragma unroll
  for (int n = 0; n < DSTATE; ++n) Arow[n] = -__expf(a_log[d * DSTATE + n]);
  float Dd = dskip[d];
  float h[DSTATE];
  #pragma unroll
  for (int n = 0; n < DSTATE; ++n) h[n] = 0.f;

  for (int t = 0; t < LSEQ; ++t) {
    size_t m = (size_t)b * LSEQ + t;
    float dlt = delta[m * DINNER + d];
    float u   = bf2f(xc[m * DINNER + d]);
    float du  = dlt * u;
    const float* bc = xdbl + m * 64;     // [0:32)=dt, [32:48)=B, [48:64)=C
    float y = 0.f;
    #pragma unroll
    for (int n = 0; n < DSTATE; ++n) {
      h[n] = h[n] * __expf(dlt * Arow[n]) + du * bc[32 + n];
      y = fmaf(h[n], bc[48 + n], y);
    }
    float z = bf2f(xz[m * (2 * DINNER) + DINNER + d]);
    float g = z / (1.f + __expf(-z));
    yg[m * DINNER + d] = f2bf((y + u * Dd) * g);
  }
}

// ---------------- prep kernels ----------------
__global__ void cvt_f32_bf16(const float* __restrict__ src,
                             unsigned short* __restrict__ dst, int n)
{
  int i = blockIdx.x * blockDim.x + threadIdx.x;
  if (i < n) dst[i] = f2bf(src[i]);
}

// (O, I, 3, 3) f32 -> (O, p=kh*3+kw, I) bf16
__global__ void reorder_conv_w(const float* __restrict__ src,
                               unsigned short* __restrict__ dst)
{
  int idx = blockIdx.x * blockDim.x + threadIdx.x;
  if (idx >= CDIM * 9 * CDIM) return;
  int o  = idx / (9 * CDIM);
  int r  = idx % (9 * CDIM);
  int p  = r / CDIM;
  int ci = r % CDIM;
  dst[idx] = f2bf(src[((size_t)o * CDIM + ci) * 9 + p]);
}

// Fill zero-padded bf16 NHWC (B,26,26,512) from either NCHW f32 (mode 0)
// or the (b,l,c) f32 sequence buffer (mode 1, fully coalesced in c).
__global__ void pad_fill(const float* __restrict__ xnchw,
                         const float* __restrict__ seq,
                         int mode, unsigned short* __restrict__ dst)
{
  int idx = blockIdx.x * blockDim.x + threadIdx.x;
  if (idx >= BDIM * PADAREA * CDIM) return;
  int c    = idx & (CDIM - 1);
  int rest = idx >> 9;               // (b*26 + ph)*26 + pw
  int pw   = rest % PADHW;
  int ph   = (rest / PADHW) % PADHW;
  int b    = rest / PADAREA;
  float v = 0.f;
  if (ph >= 1 && ph <= HH && pw >= 1 && pw <= WWDIM) {
    int h = ph - 1, w = pw - 1;
    if (mode == 0) v = xnchw[(((size_t)b * CDIM + c) * HH + h) * WWDIM + w];
    else           v = seq[((size_t)b * LSEQ + h * WWDIM + w) * CDIM + c];
  }
  dst[idx] = f2bf(v);
}

// ---------------- host launcher ----------------
extern "C" void kernel_launch(void* const* d_in, const int* in_sizes, int n_in,
                              void* d_out, int out_size, void* d_ws, size_t ws_size,
                              hipStream_t stream) {
  (void)in_sizes; (void)n_in; (void)out_size; (void)ws_size;
  const float* x          = (const float*)d_in[0];
  const float* conv_in_w  = (const float*)d_in[1];
  const float* conv_in_b  = (const float*)d_in[2];
  const float* bn_in_g    = (const float*)d_in[3];
  const float* bn_in_b    = (const float*)d_in[4];
  const float* norm_w     = (const float*)d_in[5];
  const float* norm_b     = (const float*)d_in[6];
  const float* in_w       = (const float*)d_in[7];
  const float* conv1d_w   = (const float*)d_in[8];
  const float* conv1d_b   = (const float*)d_in[9];
  const float* xproj_w    = (const float*)d_in[10];
  const float* dtproj_w   = (const float*)d_in[11];
  const float* dtproj_b   = (const float*)d_in[12];
  const float* a_log      = (const float*)d_in[13];
  const float* d_skip     = (const float*)d_in[14];
  const float* out_w      = (const float*)d_in[15];
  const float* conv_out_w = (const float*)d_in[16];
  const float* conv_out_b = (const float*)d_in[17];
  const float* bn_out_g   = (const float*)d_in[18];
  const float* bn_out_b   = (const float*)d_in[19];
  float* out = (float*)d_out;

  char* ws = (char*)d_ws;
  size_t off = 0;
  auto take = [&](size_t elems, size_t esz) -> void* {
    void* p = ws + off;
    off += ((elems * esz + 255) & ~(size_t)255);
    return p;
  };
  unsigned short* w_cin  = (unsigned short*)take(CDIM * 9 * CDIM, 2);          // conv_in reordered
  unsigned short* w_cout = (unsigned short*)take(CDIM * 9 * CDIM, 2);          // conv_out reordered
  unsigned short* w_in   = (unsigned short*)take(2 * 2048 * CDIM, 2);          // in_proj
  unsigned short* w_xp   = (unsigned short*)take(2 * 64 * DINNER, 2);          // x_proj
  unsigned short* w_dt   = (unsigned short*)take(2 * DINNER * DTRANK, 2);      // dt_proj
  unsigned short* w_out  = (unsigned short*)take(2 * CDIM * DINNER, 2);        // out_proj
  unsigned short* pad    = (unsigned short*)take((size_t)BDIM * PADAREA * CDIM, 2);
  float*          seq    = (float*)take((size_t)MROWS * CDIM, 4);              // residual stream
  unsigned short* hbf    = (unsigned short*)take((size_t)MROWS * CDIM, 2);     // post-LN
  unsigned short* xz     = (unsigned short*)take((size_t)MROWS * 2 * DINNER, 2);
  unsigned short* xc     = (unsigned short*)take((size_t)MROWS * DINNER, 2);
  float*          xdbl   = (float*)take((size_t)MROWS * 64, 4);
  unsigned short* dtbf   = (unsigned short*)take((size_t)MROWS * DTRANK, 2);
  float*          delta  = (float*)take((size_t)MROWS * DINNER, 4);
  unsigned short* yg     = (unsigned short*)take((size_t)MROWS * DINNER, 2);

  const int TPB = 256;
  auto blks = [](size_t n, int t) { return (unsigned)((n + t - 1) / t); };

  // --- weight prep ---
  reorder_conv_w<<<blks((size_t)CDIM * 9 * CDIM, TPB), TPB, 0, stream>>>(conv_in_w, w_cin);
  reorder_conv_w<<<blks((size_t)CDIM * 9 * CDIM, TPB), TPB, 0, stream>>>(conv_out_w, w_cout);
  cvt_f32_bf16<<<blks(2 * 2048 * CDIM, TPB), TPB, 0, stream>>>(in_w, w_in, 2 * 2048 * CDIM);
  cvt_f32_bf16<<<blks(2 * 64 * DINNER, TPB), TPB, 0, stream>>>(xproj_w, w_xp, 2 * 64 * DINNER);
  cvt_f32_bf16<<<blks(2 * DINNER * DTRANK, TPB), TPB, 0, stream>>>(dtproj_w, w_dt, 2 * DINNER * DTRANK);
  cvt_f32_bf16<<<blks(2 * CDIM * DINNER, TPB), TPB, 0, stream>>>(out_w, w_out, 2 * CDIM * DINNER);

  // --- conv_in + BN + ReLU -> seq (b,l,c) f32 ---
  pad_fill<<<blks((size_t)BDIM * PADAREA * CDIM, TPB), TPB, 0, stream>>>(x, nullptr, 0, pad);
  dim3 gc(CDIM / 64, MROWS / 32);
  conv3x3_gemm_wmma<<<gc, 32, 0, stream>>>(pad, w_cin, EPI_CONV_IN,
                                           conv_in_b, bn_in_g, bn_in_b, seq);

  // --- Mamba blocks ---
  for (int i = 0; i < 2; ++i) {
    layernorm_rows<<<MROWS, 32, 0, stream>>>(seq, norm_w + i * CDIM, norm_b + i * CDIM, hbf);

    dim3 g1(2048 / 64, MROWS / 32);
    gemm_bf16_wmma<<<g1, 32, 0, stream>>>(hbf, w_in + (size_t)i * 2048 * CDIM,
                                          MROWS, 2048, CDIM, EPI_BF16,
                                          nullptr, nullptr, nullptr, nullptr, xz);

    dwconv1d_silu<<<blks((size_t)MROWS * DINNER, TPB), TPB, 0, stream>>>(
        xz, conv1d_w + (size_t)i * DINNER * 4, conv1d_b + (size_t)i * DINNER, xc);

    dim3 g2(64 / 64, MROWS / 32);
    gemm_bf16_wmma<<<g2, 32, 0, stream>>>(xc, w_xp + (size_t)i * 64 * DINNER,
                                          MROWS, 64, DINNER, EPI_XPROJ,
                                          nullptr, nullptr, nullptr, xdbl, dtbf);

    dim3 g3(DINNER / 64, MROWS / 32);
    gemm_bf16_wmma<<<g3, 32, 0, stream>>>(dtbf, w_dt + (size_t)i * DINNER * DTRANK,
                                          MROWS, DINNER, DTRANK, EPI_SOFTPLUS,
                                          dtproj_b + (size_t)i * DINNER, nullptr, nullptr,
                                          delta, nullptr);

    selective_scan_fused<<<blks(BDIM * DINNER, TPB), TPB, 0, stream>>>(
        delta, xc, xdbl, xz,
        a_log + (size_t)i * DINNER * DSTATE, d_skip + (size_t)i * DINNER, yg);

    dim3 g4(CDIM / 64, MROWS / 32);
    gemm_bf16_wmma<<<g4, 32, 0, stream>>>(yg, w_out + (size_t)i * CDIM * DINNER,
                                          MROWS, CDIM, DINNER, EPI_RESID,
                                          nullptr, nullptr, nullptr, seq, nullptr);
  }

  // --- conv_out + BN + ReLU -> d_out NCHW f32 ---
  pad_fill<<<blks((size_t)BDIM * PADAREA * CDIM, TPB), TPB, 0, stream>>>(nullptr, seq, 1, pad);
  conv3x3_gemm_wmma<<<gc, 32, 0, stream>>>(pad, w_cout, EPI_CONV_OUT,
                                           conv_out_b, bn_out_g, bn_out_b, out);
}
